// EncModule_25022479466835
// MI455X (gfx1250) — compile-verified, matrix operand order checked by reading, not run on hardware
//
#include <hip/hip_runtime.h>
#include <hip/hip_bf16.h>

#define B_DIM 16
#define C_DIM 512
#define N_DIM 4096
#define K_CW  32
#define EPSV  1e-5f
#define LDA   40   // padded LDS row stride (bf16) for 32-wide tiles: 80B, 20-bank offset
#define LDP   72   // padded LDS row stride (bf16) for 64-wide tiles: 144B (128B data + 16B pad)

typedef __attribute__((ext_vector_type(16))) __bf16 v16bf;
typedef __attribute__((ext_vector_type(8)))  float  v8f;
typedef __attribute__((ext_vector_type(4)))  unsigned int v4u;
typedef __attribute__((ext_vector_type(8)))  int v8i;
typedef __attribute__((ext_vector_type(4)))  int v4i;

struct alignas(16) U128 { unsigned int x, y, z, w; };
union FragBF { v16bf v; U128 q[2]; };

__device__ __forceinline__ void split_bf16(float f, __bf16& hi, __bf16& lo) {
  __bf16 h = (__bf16)f;
  hi = h;
  lo = (__bf16)(f - (float)h);
}

// A/B fragment for v_wmma_*_16x16x32: lane holds row/col = lane&15,
// K chunks [ksel*8 .. +7] and [16+ksel*8 .. +7]  (ksel = lane>>4)
__device__ __forceinline__ FragBF load_frag(const __bf16* rowptr, int ksel) {
  FragBF f;
  f.q[0] = *reinterpret_cast<const U128*>(rowptr + ksel * 8);
  f.q[1] = *reinterpret_cast<const U128*>(rowptr + 16 + ksel * 8);
  return f;
}

// fp32-ish GEMM via bf16 hi/lo split: a*b ~= aH*bH + aH*bL + aL*bH
__device__ __forceinline__ v8f wmma3(const FragBF& aH, const FragBF& aL,
                                     const FragBF& bH, const FragBF& bL, v8f c) {
  c = __builtin_amdgcn_wmma_f32_16x16x32_bf16(false, aH.v, false, bH.v, (short)0, c, false, false);
  c = __builtin_amdgcn_wmma_f32_16x16x32_bf16(false, aH.v, false, bL.v, (short)0, c, false, false);
  c = __builtin_amdgcn_wmma_f32_16x16x32_bf16(false, aL.v, false, bH.v, (short)0, c, false, false);
  return c;
}

// per-lane async global->LDS 16B copy (ASYNCcnt tracked)
__device__ __forceinline__ void async_b128(unsigned lds_addr, const void* gaddr) {
  asm volatile("global_load_async_to_lds_b128 %0, %1, off"
               :: "v"(lds_addr), "v"(gaddr) : "memory");
}
__device__ __forceinline__ void wait_async0() {
  asm volatile("s_wait_asynccnt 0x0" ::: "memory");
}
__device__ __forceinline__ unsigned lds_off(const void* p) {
  return (unsigned)(size_t)p;   // generic LDS addr = {aperture, offset[31:0]}
}

// ---------------------------------------------------------------------------
// Kernel 1: h[b] = relu(BN2(conv_w @ x[b])), stored as separate bf16 hi/lo
// planes (hi = bf16(v), lo = bf16(v-hi)) so downstream kernels can DMA them.
// 128x128 tile per block, k-step 32, 8 waves x 8 accum tiles.
// ---------------------------------------------------------------------------
__global__ __launch_bounds__(256) void k_conv_bn_relu(
    const float* __restrict__ x, const float* __restrict__ w,
    const float* __restrict__ bn_w, const float* __restrict__ bn_b,
    const float* __restrict__ bn_m, const float* __restrict__ bn_v,
    __bf16* __restrict__ h_hi, __bf16* __restrict__ h_lo)
{
  __shared__ alignas(16) __bf16 As_hi[128][LDA], As_lo[128][LDA];
  __shared__ alignas(16) __bf16 Bs_hi[128][LDA], Bs_lo[128][LDA];

  const int b  = blockIdx.z;
  const int c0 = blockIdx.y * 128;
  const int n0 = blockIdx.x * 128;
  const int t    = threadIdx.x;
  const int wave = t >> 5;
  const int lane = t & 31;
  const int lrow = lane & 15;
  const int ksel = lane >> 4;

  v8f acc[8];
#pragma unroll
  for (int j = 0; j < 8; ++j)
#pragma unroll
    for (int e = 0; e < 8; ++e) acc[j][e] = 0.f;

  const float* xb = x + (size_t)b * C_DIM * N_DIM;

  for (int k0 = 0; k0 < C_DIM; k0 += 32) {
#pragma unroll
    for (int i = 0; i < 16; ++i) {
      int l = t + i * 256;
      int r = l >> 5, cI = l & 31;
      split_bf16(w[(c0 + r) * C_DIM + k0 + cI], As_hi[r][cI], As_lo[r][cI]);
    }
#pragma unroll
    for (int i = 0; i < 16; ++i) {
      int l = t + i * 256;
      int kk = l >> 7, cI = l & 127;
      split_bf16(xb[(size_t)(k0 + kk) * N_DIM + n0 + cI], Bs_hi[cI][kk], Bs_lo[cI][kk]);
    }
    __syncthreads();

    const int ar = wave * 16 + lrow;
    FragBF aH = load_frag(&As_hi[ar][0], ksel);
    FragBF aL = load_frag(&As_lo[ar][0], ksel);
#pragma unroll
    for (int j = 0; j < 8; ++j) {
      const int br = j * 16 + lrow;
      FragBF bH = load_frag(&Bs_hi[br][0], ksel);
      FragBF bL = load_frag(&Bs_lo[br][0], ksel);
      acc[j] = wmma3(aH, aL, bH, bL, acc[j]);
    }
    __syncthreads();
  }

  // epilogue: BN2 + ReLU, split to bf16 hi/lo planes
  // C/D layout: element (M = v + 8*ksel, N = lrow) in acc[j][v], tile (wave, j)
#pragma unroll
  for (int v = 0; v < 8; ++v) {
    int c = c0 + wave * 16 + ksel * 8 + v;
    float s  = bn_w[c] * rsqrtf(bn_v[c] + EPSV);
    float sh = bn_b[c] - bn_m[c] * s;
    size_t base = ((size_t)b * C_DIM + c) * N_DIM + n0 + lrow;
#pragma unroll
    for (int j = 0; j < 8; ++j) {
      float val = fmaf(acc[j][v], s, sh);
      val = val > 0.f ? val : 0.f;
      __bf16 hi, lo;
      split_bf16(val, hi, lo);
      h_hi[base + j * 16] = hi;
      h_lo[base + j * 16] = lo;
    }
  }
}

// ---------------------------------------------------------------------------
// Kernel 2: encoding. Per block: batch b, 64-position chunk.
//  phase1: xc[pos][k] = X . cw^T (WMMA), x_sq from exact hi+lo reconstruction
//  softmax over K=32 -> A (bf16 hi/lo, A^T layout in LDS)
//  phase2: E_part[k][d] = A^T . X (WMMA); X staged by TDM (hi plane) +
//          global_load_async_to_lds_b128 (lo plane), zero VALU staging.
// ---------------------------------------------------------------------------
__global__ __launch_bounds__(256) void k_encode(
    const __bf16* __restrict__ h_hi, const __bf16* __restrict__ h_lo,
    const float* __restrict__ cw,
    const float* __restrict__ scl, const float* __restrict__ c_sq,
    float* __restrict__ E_g, float* __restrict__ asum_g)
{
  __shared__ alignas(16) __bf16 Xs_hi[64][LDA], Xs_lo[64][LDA];
  __shared__ alignas(16) __bf16 Cs_hi[32][LDA], Cs_lo[32][LDA];
  __shared__ alignas(16) __bf16 At_hi[32][LDP], At_lo[32][LDP];
  __shared__ alignas(16) __bf16 X2_hi[64][LDP], X2_lo[64][LDP];
  __shared__ float  xcs[64][33];
  __shared__ float  xsq[64];
  __shared__ float  asum_l[32];

  const int b  = blockIdx.y;
  const int n0 = blockIdx.x * 64;
  const int t    = threadIdx.x;
  const int wave = t >> 5;
  const int lane = t & 31;
  const int lrow = lane & 15;
  const int ksel = lane >> 4;

  if (t < 64) xsq[t] = 0.f;
  if (t < 32) asum_l[t] = 0.f;

  const __bf16* phi = h_hi + (size_t)b * C_DIM * N_DIM;
  const __bf16* plo = h_lo + (size_t)b * C_DIM * N_DIM;

  // ---- phase 1: xc = X . cw^T  (M=64 pos, N=32 k, inner d=512) ----
  v8f accx;
#pragma unroll
  for (int e = 0; e < 8; ++e) accx[e] = 0.f;
  float xsq_part = 0.f;
  const int rt = wave >> 1;   // position tile 0..3
  const int ct = wave & 1;    // codeword tile 0..1

  for (int d0 = 0; d0 < C_DIM; d0 += 32) {
#pragma unroll
    for (int i = 0; i < 8; ++i) {
      int l = t + i * 256;
      int dl = l >> 6, p = l & 63;             // p == t&63 for all i
      size_t gi = (size_t)(d0 + dl) * N_DIM + n0 + p;
      __bf16 hi = phi[gi], lo = plo[gi];
      float v = (float)hi + (float)lo;         // exact reconstruction
      xsq_part = fmaf(v, v, xsq_part);
      Xs_hi[p][dl] = hi;
      Xs_lo[p][dl] = lo;
    }
#pragma unroll
    for (int i = 0; i < 4; ++i) {
      int l = t + i * 256;
      int kk = l >> 5, dl = l & 31;
      split_bf16(cw[kk * C_DIM + d0 + dl], Cs_hi[kk][dl], Cs_lo[kk][dl]);
    }
    __syncthreads();
    FragBF aH = load_frag(&Xs_hi[rt * 16 + lrow][0], ksel);
    FragBF aL = load_frag(&Xs_lo[rt * 16 + lrow][0], ksel);
    FragBF bH = load_frag(&Cs_hi[ct * 16 + lrow][0], ksel);
    FragBF bL = load_frag(&Cs_lo[ct * 16 + lrow][0], ksel);
    accx = wmma3(aH, aL, bH, bL, accx);
    __syncthreads();
  }
  atomicAdd(&xsq[t & 63], xsq_part);
#pragma unroll
  for (int v = 0; v < 8; ++v)
    xcs[rt * 16 + ksel * 8 + v][ct * 16 + lrow] = accx[v];
  __syncthreads();

  // ---- softmax over K per position ----
  if (t < 64) {
    const int p = t;
    float xq = xsq[p];
    float sl[K_CW];
    float m = -1e30f;
#pragma unroll
    for (int k = 0; k < K_CW; ++k) {
      float v = scl[k] * (xq - 2.f * xcs[p][k] + c_sq[k]);
      sl[k] = v;
      m = fmaxf(m, v);
    }
    float sum = 0.f;
#pragma unroll
    for (int k = 0; k < K_CW; ++k) { sl[k] = __expf(sl[k] - m); sum += sl[k]; }
    float inv = 1.f / sum;
#pragma unroll
    for (int k = 0; k < K_CW; ++k) {
      float a = sl[k] * inv;
      split_bf16(a, At_hi[k][p], At_lo[k][p]);   // A^T layout [k][pos]
      atomicAdd(&asum_l[k], a);
    }
  }
  __syncthreads();
  if (t < 32) atomicAdd(&asum_g[b * K_CW + t], asum_l[t]);

  // ---- phase 2: E_part[k][d] = A^T . X  (M=32 k, inner pos=64, N=512 d) ----
  const int mt = wave >> 2;   // k tile 0..1
  const int nt = wave & 3;    // d subtile 0..3 within a 64-col slab
  for (int d0 = 0; d0 < C_DIM; d0 += 64) {
    // hi plane: one TDM descriptor moves the whole 64x64-elem tile
    // (data_size=2B, tile_dim0=64 elem=128B rows, row stride N, LDS pad:
    //  pad_interval code 4 = every 128B, pad_amount code 3 = 16B -> 144B rows)
    if (wave == 0) {
      unsigned long long ga =
          (unsigned long long)(size_t)(const void*)(phi + (size_t)d0 * N_DIM + n0);
      v4u g0;
      g0.x = 1u;                                            // count=1
      g0.y = lds_off(&X2_hi[0][0]);                          // lds_addr
      g0.z = (unsigned)(ga & 0xFFFFFFFFu);                   // global_addr lo
      g0.w = (unsigned)((ga >> 32) & 0x01FFFFFFu) | (2u << 30); // addr hi | type=2
      v8i g1;
      g1[0] = (int)((1u << 16) | (1u << 20) | (4u << 22) | (3u << 25));
      g1[1] = (int)(((unsigned)N_DIM & 0xFFFFu) << 16);      // tensor_dim0[15:0]
      g1[2] = (int)((((unsigned)N_DIM >> 16) & 0xFFFFu) |
                    (((unsigned)C_DIM & 0xFFFFu) << 16));    // dim0 hi | dim1 lo
      g1[3] = (int)(((unsigned)C_DIM >> 16) | (64u << 16));  // dim1 hi | tile_dim0
      g1[4] = (int)64;                                       // tile_dim1
      g1[5] = (int)N_DIM;                                    // dim0_stride[31:0]
      g1[6] = 0;
      g1[7] = 0;
      v4i gz4 = {0, 0, 0, 0};
      v8i gz8 = {0, 0, 0, 0, 0, 0, 0, 0};
      __builtin_amdgcn_tensor_load_to_lds(g0, g1, gz4, gz4, gz8, 0);
    }
    // lo plane: per-lane 16B async copies (64 rows x 8 chunks = 512 ops)
#pragma unroll
    for (int i = 0; i < 2; ++i) {
      int ci = t + i * 256;
      int dl = ci >> 3, ch = ci & 7;
      const void* g = (const void*)(plo + (size_t)(d0 + dl) * N_DIM + n0 + ch * 8);
      async_b128(lds_off(&X2_lo[dl][ch * 8]), g);
    }
    wait_async0();
    if (wave == 0) __builtin_amdgcn_s_wait_tensorcnt(0);
    __syncthreads();

    v8f accE;
#pragma unroll
    for (int e = 0; e < 8; ++e) accE[e] = 0.f;
#pragma unroll
    for (int p0 = 0; p0 < 64; p0 += 32) {
      FragBF aH = load_frag(&At_hi[mt * 16 + lrow][p0], ksel);
      FragBF aL = load_frag(&At_lo[mt * 16 + lrow][p0], ksel);
      FragBF bH = load_frag(&X2_hi[nt * 16 + lrow][p0], ksel);
      FragBF bL = load_frag(&X2_lo[nt * 16 + lrow][p0], ksel);
      accE = wmma3(aH, aL, bH, bL, accE);
    }
#pragma unroll
    for (int v = 0; v < 8; ++v) {
      int k = mt * 16 + ksel * 8 + v;
      int d = d0 + nt * 16 + lrow;
      atomicAdd(&E_g[((size_t)b * K_CW + k) * C_DIM + d], accE[v]);
    }
    __syncthreads();
  }
}

// ---------------------------------------------------------------------------
// Kernel 3: en[b,d] = mean_k relu(BN1(E[b,k,d] - asum[b,k]*cw[k,d]))
// ---------------------------------------------------------------------------
__global__ __launch_bounds__(256) void k_en(
    const float* __restrict__ E_g, const float* __restrict__ asum_g,
    const float* __restrict__ cw,
    const float* __restrict__ b1w, const float* __restrict__ b1b,
    const float* __restrict__ b1m, const float* __restrict__ b1v,
    float* __restrict__ en_g)
{
  int idx = blockIdx.x * 256 + threadIdx.x;     // B*C = 8192
  if (idx >= B_DIM * C_DIM) return;
  int b = idx >> 9, d = idx & (C_DIM - 1);
  float sum = 0.f;
#pragma unroll 4
  for (int k = 0; k < K_CW; ++k) {
    float e = E_g[((size_t)b * K_CW + k) * C_DIM + d]
            - asum_g[b * K_CW + k] * cw[k * C_DIM + d];
    float s  = b1w[k] * rsqrtf(b1v[k] + EPSV);
    float sh = b1b[k] - b1m[k] * s;
    float v = fmaf(e, s, sh);
    sum += fmaxf(v, 0.f);
  }
  en_g[idx] = sum * (1.f / K_CW);
}

// ---------------------------------------------------------------------------
// Kernel 4: gamma = sigmoid(en @ fc_w^T + fc_b); se_out = en @ se_w^T + se_b
// ---------------------------------------------------------------------------
__global__ __launch_bounds__(512) void k_gate(
    const float* __restrict__ en_g, const float* __restrict__ fc_w,
    const float* __restrict__ fc_b, const float* __restrict__ se_w,
    const float* __restrict__ se_b, float* __restrict__ gamma_g,
    float* __restrict__ se_out)
{
  __shared__ float red[512];
  const int b = blockIdx.x, o = threadIdx.x;
  const float* en = en_g + b * C_DIM;
  float acc = 0.f;
  for (int d = 0; d < C_DIM; ++d) acc = fmaf(en[d], fc_w[o * C_DIM + d], acc);
  acc += fc_b[o];
  gamma_g[b * C_DIM + o] = 1.f / (1.f + __expf(-acc));
  red[o] = en[o] * se_w[o];
  __syncthreads();
  for (int s = 256; s > 0; s >>= 1) {
    if (o < s) red[o] += red[o + s];
    __syncthreads();
  }
  if (o == 0) se_out[b] = red[0] + se_b[0];
}

// ---------------------------------------------------------------------------
// Kernel 5: out = relu(x * (1 + gamma[b,c]))   (float4, pure bandwidth)
// ---------------------------------------------------------------------------
__global__ __launch_bounds__(256) void k_out(
    const float* __restrict__ x, const float* __restrict__ gamma_g,
    float* __restrict__ out)
{
  size_t i4 = (size_t)blockIdx.x * 256 + threadIdx.x;
  size_t i  = i4 * 4;
  int bc = (int)(i >> 12);                      // 4096 spatial per (b,c)
  float g1 = 1.f + gamma_g[bc];
  float4 xv = *reinterpret_cast<const float4*>(x + i);
  float4 o;
  o.x = fmaxf(xv.x * g1, 0.f);
  o.y = fmaxf(xv.y * g1, 0.f);
  o.z = fmaxf(xv.z * g1, 0.f);
  o.w = fmaxf(xv.w * g1, 0.f);
  *reinterpret_cast<float4*>(out + i) = o;
}

__global__ void k_zero(float* __restrict__ p, int n) {
  int i = blockIdx.x * 256 + threadIdx.x;
  if (i < n) p[i] = 0.f;
}

__global__ void k_csq(const float* __restrict__ cw, float* __restrict__ c_sq) {
  int k = threadIdx.x;
  if (k < K_CW) {
    float s = 0.f;
    for (int d = 0; d < C_DIM; ++d) s = fmaf(cw[k * C_DIM + d], cw[k * C_DIM + d], s);
    c_sq[k] = s;
  }
}

extern "C" void kernel_launch(void* const* d_in, const int* in_sizes, int n_in,
                              void* d_out, int out_size, void* d_ws, size_t ws_size,
                              hipStream_t stream) {
  (void)in_sizes; (void)n_in; (void)out_size; (void)ws_size;
  const float* x      = (const float*)d_in[0];
  const float* conv_w = (const float*)d_in[1];
  const float* bn2_w  = (const float*)d_in[2];
  const float* bn2_b  = (const float*)d_in[3];
  const float* bn2_m  = (const float*)d_in[4];
  const float* bn2_v  = (const float*)d_in[5];
  const float* cw     = (const float*)d_in[6];
  const float* scl    = (const float*)d_in[7];
  const float* bn1_w  = (const float*)d_in[8];
  const float* bn1_b  = (const float*)d_in[9];
  const float* bn1_m  = (const float*)d_in[10];
  const float* bn1_v  = (const float*)d_in[11];
  const float* fc_w   = (const float*)d_in[12];
  const float* fc_b   = (const float*)d_in[13];
  const float* se_w   = (const float*)d_in[14];
  const float* se_b   = (const float*)d_in[15];

  float* out    = (float*)d_out;
  float* se_out = out + (size_t)B_DIM * C_DIM * N_DIM;

  const size_t HB = (size_t)B_DIM * C_DIM * N_DIM * sizeof(__bf16);  // 64 MB/plane
  char* wsb = (char*)d_ws;
  __bf16* h_hi  = (__bf16*)(wsb);
  __bf16* h_lo  = (__bf16*)(wsb + HB);
  float* E_g    = (float*)(wsb + 2 * HB);
  float* asum_g = E_g + (size_t)B_DIM * K_CW * C_DIM;
  float* en_g   = asum_g + B_DIM * K_CW;
  float* gamma  = en_g + B_DIM * C_DIM;
  float* c_sq   = gamma + B_DIM * C_DIM;

  // E_g and asum_g are contiguous -> one zeroing pass (atomics target)
  int nz = B_DIM * K_CW * C_DIM + B_DIM * K_CW;
  k_zero<<<(nz + 255) / 256, 256, 0, stream>>>(E_g, nz);
  k_csq<<<1, 32, 0, stream>>>(cw, c_sq);

  k_conv_bn_relu<<<dim3(N_DIM / 128, C_DIM / 128, B_DIM), 256, 0, stream>>>(
      x, conv_w, bn2_w, bn2_b, bn2_m, bn2_v, h_hi, h_lo);

  k_encode<<<dim3(N_DIM / 64, B_DIM), 256, 0, stream>>>(
      h_hi, h_lo, cw, scl, c_sq, E_g, asum_g);

  k_en<<<(B_DIM * C_DIM + 255) / 256, 256, 0, stream>>>(
      E_g, asum_g, cw, bn1_w, bn1_b, bn1_m, bn1_v, en_g);

  k_gate<<<B_DIM, 512, 0, stream>>>(en_g, fc_w, fc_b, se_w, se_b, gamma, se_out);

  k_out<<<(int)((size_t)B_DIM * C_DIM * N_DIM / 4 / 256), 256, 0, stream>>>(
      x, gamma, out);
}
